// CausalSelfAttention_45114336477749
// MI455X (gfx1250) — matrix operand order repaired and checked by reading, not compiled, not run on hardware
//
#include <hip/hip_runtime.h>
#include <hip/hip_bf16.h>

#define D_MODEL 1024
#define N_HEADS 16
#define HEAD_DIM 64
#define SEQ 2048

typedef __attribute__((ext_vector_type(16))) __bf16 bf16x16;
typedef __attribute__((ext_vector_type(8)))  float  f32x8;

union Frag16 { bf16x16 v; uint4 q[2]; };

__device__ __forceinline__ unsigned short f2bf(float f) {
  unsigned u = __builtin_bit_cast(unsigned, f);
  u += 0x7FFFu + ((u >> 16) & 1u);        // round-to-nearest-even
  return (unsigned short)(u >> 16);
}

__device__ __forceinline__ f32x8 zero8() {
  f32x8 z;
#pragma unroll
  for (int i = 0; i < 8; ++i) z[i] = 0.f;
  return z;
}

// ---- CDNA5 async copy: 16B/lane global -> LDS, tracked by ASYNCcnt --------
__device__ __forceinline__ void async_cp16(const unsigned short* g,
                                           unsigned short* l) {
  // generic LDS pointer: addr[31:0] is the LDS byte offset (ISA 10.2)
  unsigned lds_off = (unsigned)(unsigned long long)l;
  asm volatile("global_load_async_to_lds_b128 %0, %1, off"
               :: "v"(lds_off), "v"(g) : "memory");
}
__device__ __forceinline__ void wait_async_le4() {
  asm volatile("s_wait_asynccnt 4" ::: "memory");
}
__device__ __forceinline__ void wait_async_0() {
  asm volatile("s_wait_asynccnt 0" ::: "memory");
}

// -------------------------------------------------------------------------
// Generic 128x128x32 bf16-WMMA GEMM.
// MODE 0: X[4096,1024] x {Wq|Wk|Wv}[1024,1024] + b -> Q/K (bf16 [B,H,S,64]),
//         V transposed (bf16 [B,H,64,S]).  grid.y covers N=3072.
// MODE 1: X[4096,1024] x Wo + bo -> fp32 out [4096,1024].
// -------------------------------------------------------------------------
template <int MODE>
__global__ __launch_bounds__(256)
void gemm_kernel(const float* __restrict__ X,
                 const float* __restrict__ W0, const float* __restrict__ W1,
                 const float* __restrict__ W2,
                 const float* __restrict__ b0, const float* __restrict__ b1,
                 const float* __restrict__ b2,
                 unsigned short* __restrict__ Qb,
                 unsigned short* __restrict__ Kb,
                 unsigned short* __restrict__ Vt,
                 float* __restrict__ OutF)
{
  constexpr int BM = 128, BN = 128, BK = 32, LDA = 40;
  __shared__ alignas(16) unsigned short As[BM * LDA];        // [m][k]
  __shared__ alignas(16) unsigned short Bs[BN * LDA];        // [n][k] (transposed)

  const int tid  = threadIdx.x;
  const int lane = tid & 31;
  const int w    = tid >> 5;            // 8 waves
  const int wm   = w & 3, wn = w >> 2;  // 4x2 wave grid -> 32x64 per wave
  const int m16  = lane & 15;
  const int half = lane >> 4;

  const int m0  = blockIdx.x * BM;
  const int n0g = blockIdx.y * BN;
  int sel = 0, n0 = n0g;
  const float* W = W0; const float* bias = b0;
  if (MODE == 0) {
    sel = n0g >> 10; n0 = n0g & 1023;
    W    = (sel == 0) ? W0 : (sel == 1) ? W1 : W2;
    bias = (sel == 0) ? b0 : (sel == 1) ? b1 : b2;
  }

  f32x8 acc[2][4];
#pragma unroll
  for (int i = 0; i < 2; ++i)
#pragma unroll
    for (int j = 0; j < 4; ++j) acc[i][j] = zero8();

  const int ra  = tid >> 3;
  const int ca4 = (tid & 7) << 2;
  const int kb8 = tid >> 5;
  const int cb4 = (tid & 31) << 2;

  for (int k0 = 0; k0 < 1024; k0 += BK) {
    // prefetch next k-step tiles into L2 while this one is consumed
    if (k0 + BK < 1024) {
      __builtin_prefetch(X + (size_t)(m0 + ra) * 1024 + k0 + BK + ca4, 0, 3);
      __builtin_prefetch(W + (size_t)(k0 + BK + kb8) * 1024 + n0 + cb4, 0, 3);
    }
    // ---- stage A tile (fp32 -> bf16), coalesced float4 loads ----
#pragma unroll
    for (int rr = 0; rr < BM; rr += 32) {
      float4 f = *(const float4*)(X + (size_t)(m0 + ra + rr) * 1024 + k0 + ca4);
      uint2 p;
      p.x = (unsigned)f2bf(f.x) | ((unsigned)f2bf(f.y) << 16);
      p.y = (unsigned)f2bf(f.z) | ((unsigned)f2bf(f.w) << 16);
      *(uint2*)&As[(ra + rr) * LDA + ca4] = p;
    }
    // ---- stage W tile transposed ([n][k]) ----
#pragma unroll
    for (int k2 = 0; k2 < BK; k2 += 8) {
      float4 f = *(const float4*)(W + (size_t)(k0 + kb8 + k2) * 1024 + n0 + cb4);
      Bs[(cb4 + 0) * LDA + kb8 + k2] = f2bf(f.x);
      Bs[(cb4 + 1) * LDA + kb8 + k2] = f2bf(f.y);
      Bs[(cb4 + 2) * LDA + kb8 + k2] = f2bf(f.z);
      Bs[(cb4 + 3) * LDA + kb8 + k2] = f2bf(f.w);
    }
    __syncthreads();

    Frag16 a[2], b[4];
#pragma unroll
    for (int i = 0; i < 2; ++i) {           // A: lane = row, K chunks 8c / 16+8c
      const int r = wm * 32 + 16 * i + m16;
      a[i].q[0] = *(const uint4*)&As[r * LDA + 8 * half];
      a[i].q[1] = *(const uint4*)&As[r * LDA + 16 + 8 * half];
    }
#pragma unroll
    for (int j = 0; j < 4; ++j) {           // B: lane = col, K chunks 8c / 16+8c
      const int col = wn * 64 + 16 * j + m16;
      b[j].q[0] = *(const uint4*)&Bs[col * LDA + 8 * half];
      b[j].q[1] = *(const uint4*)&Bs[col * LDA + 16 + 8 * half];
    }
#pragma unroll
    for (int i = 0; i < 2; ++i)
#pragma unroll
      for (int j = 0; j < 4; ++j)
        acc[i][j] = __builtin_amdgcn_wmma_f32_16x16x32_bf16(
            false, a[i].v, false, b[j].v, (short)0, acc[i][j], false, false);
    __syncthreads();
  }

  // ---- epilogue ----
#pragma unroll
  for (int i = 0; i < 2; ++i) {
    const int mg_base = m0 + wm * 32 + 16 * i;
#pragma unroll
    for (int j = 0; j < 4; ++j) {
      const int ng = n0g + wn * 64 + 16 * j + m16;
      const int nc = (MODE == 0) ? (ng & 1023) : ng;
      const float bval = bias[nc];
#pragma unroll
      for (int v = 0; v < 8; ++v) {
        const int   mg  = mg_base + v + 8 * half;   // C layout: row = v + 8c
        const float val = acc[i][j][v] + bval;
        if (MODE == 0) {
          const int bidx = mg >> 11;                // / SEQ
          const int s    = mg & (SEQ - 1);
          const int h    = nc >> 6;
          const int dh   = nc & 63;
          const size_t bh = (size_t)(bidx * N_HEADS + h);
          if (sel == 0)       Qb[(bh * SEQ + s) * HEAD_DIM + dh] = f2bf(val);
          else if (sel == 1)  Kb[(bh * SEQ + s) * HEAD_DIM + dh] = f2bf(val);
          else                Vt[(bh * HEAD_DIM + dh) * SEQ + s] = f2bf(val);
        } else {
          OutF[(size_t)mg * 1024 + ng] = val;
        }
      }
    }
  }
}

// -------------------------------------------------------------------------
// Flash attention: block = 128 q rows, 8 waves x 16 rows.
// K/V 64-key blocks are double-buffered in LDS via global_load_async_to_lds
// (ASYNCcnt) so the next block's DMA overlaps this block's WMMA + softmax.
// -------------------------------------------------------------------------
__global__ __launch_bounds__(256)
void attn_kernel(const unsigned short* __restrict__ Qb,
                 const unsigned short* __restrict__ Kb,
                 const unsigned short* __restrict__ Vt,
                 float* __restrict__ OutAttn)
{
  constexpr int LKV = 72;                        // padded stride: conflict-free
  constexpr int LP  = 72;
  __shared__ alignas(16) unsigned short Ks[2][64 * LKV];   // [kv][dh]
  __shared__ alignas(16) unsigned short Vs[2][64 * LKV];   // [dh][kv]
  __shared__ alignas(16) unsigned short Pw[8 * 16 * LP];

  const int tid  = threadIdx.x;
  const int lane = tid & 31;
  const int w    = tid >> 5;                     // 0..7
  const int m16  = lane & 15;
  const int half = lane >> 4;

  const int h    = blockIdx.y;
  const int bidx = blockIdx.z;
  const int q0   = blockIdx.x * 128 + w * 16;

  const size_t bh = (size_t)(bidx * N_HEADS + h);
  const unsigned short* Qp = Qb + bh * SEQ * HEAD_DIM;
  const unsigned short* Kp = Kb + bh * SEQ * HEAD_DIM;
  const unsigned short* Vp = Vt + bh * HEAD_DIM * SEQ;

  // Q fragments (A layout), K-dim = 64 split into 2x32
  Frag16 aq[2];
#pragma unroll
  for (int f = 0; f < 2; ++f) {
    const unsigned short* row = Qp + (size_t)(q0 + m16) * HEAD_DIM + 32 * f;
    aq[f].q[0] = *(const uint4*)(row + 8 * half);
    aq[f].q[1] = *(const uint4*)(row + 16 + 8 * half);
  }

  f32x8 o[4];
#pragma unroll
  for (int t = 0; t < 4; ++t) o[t] = zero8();
  float rm[8], rl[8];
#pragma unroll
  for (int v = 0; v < 8; ++v) { rm[v] = -__builtin_inff(); rl[v] = 0.f; }

  unsigned short* P = Pw + w * 16 * LP;
  const float scale = 0.125f;                    // 1/sqrt(64)

  // stage kv-block 0 (4 async b128 per wave: 2 K-chunks + 2 V-chunks)
#pragma unroll
  for (int i = 0; i < 2; ++i) {
    const int chunk = tid + 256 * i;
    const int row = chunk >> 3, c8 = (chunk & 7) << 3;
    async_cp16(Kp + (size_t)row * HEAD_DIM + c8, &Ks[0][row * LKV + c8]);
    async_cp16(Vp + (size_t)row * SEQ + c8,      &Vs[0][row * LKV + c8]);
  }

  for (int blk = 0; blk < SEQ / 64; ++blk) {
    const int kv0 = blk * 64;
    const int cur = blk & 1;
    const bool has_next = (blk + 1) < (SEQ / 64);
    if (has_next) {
#pragma unroll
      for (int i = 0; i < 2; ++i) {
        const int chunk = tid + 256 * i;
        const int row = chunk >> 3, c8 = (chunk & 7) << 3;
        async_cp16(Kp + (size_t)(kv0 + 64 + row) * HEAD_DIM + c8,
                   &Ks[cur ^ 1][row * LKV + c8]);
        async_cp16(Vp + (size_t)row * SEQ + kv0 + 64 + c8,
                   &Vs[cur ^ 1][row * LKV + c8]);
      }
      wait_async_le4();        // current block's 4 copies are done
    } else {
      wait_async_0();
    }
    __syncthreads();
    const unsigned short* Kc = Ks[cur];
    const unsigned short* Vc = Vs[cur];

    // ---- 8 score WMMAs from LDS-staged K ----
    Frag16 kb[4][2];
#pragma unroll
    for (int j = 0; j < 4; ++j) {
      const unsigned short* krow = Kc + (16 * j + m16) * LKV;
#pragma unroll
      for (int f = 0; f < 2; ++f) {
        kb[j][f].q[0] = *(const uint4*)(krow + 32 * f + 8 * half);
        kb[j][f].q[1] = *(const uint4*)(krow + 32 * f + 16 + 8 * half);
      }
    }
    f32x8 s[4];
#pragma unroll
    for (int j = 0; j < 4; ++j) {
      s[j] = __builtin_amdgcn_wmma_f32_16x16x32_bf16(
          false, aq[0].v, false, kb[j][0].v, (short)0, zero8(), false, false);
      s[j] = __builtin_amdgcn_wmma_f32_16x16x32_bf16(
          false, aq[1].v, false, kb[j][1].v, (short)0, s[j], false, false);
    }

    // ---- V fragments (LDS), issued before softmax math ----
    Frag16 vb[2][4];
#pragma unroll
    for (int f = 0; f < 2; ++f)
#pragma unroll
      for (int t = 0; t < 4; ++t) {
        const unsigned short* vrow = Vc + (16 * t + m16) * LKV + 32 * f + 8 * half;
        vb[f][t].q[0] = *(const uint4*)(vrow);
        vb[f][t].q[1] = *(const uint4*)(vrow + 16);
      }

    // ---- online softmax (row stats live in 16-lane groups) ----
    float sf[8];
#pragma unroll
    for (int v = 0; v < 8; ++v) {
      float mx = fmaxf(fmaxf(s[0][v], s[1][v]), fmaxf(s[2][v], s[3][v]));
#pragma unroll
      for (int off = 1; off < 16; off <<= 1)
        mx = fmaxf(mx, __shfl_xor(mx, off, 32));
      mx *= scale;
      const float newm = fmaxf(rm[v], mx);
      sf[v] = __expf(rm[v] - newm);
      float rs = 0.f;
#pragma unroll
      for (int j = 0; j < 4; ++j) {
        const float p = __expf(s[j][v] * scale - newm);
        rs += p;
        P[(v + 8 * half) * LP + 16 * j + m16] = f2bf(p);   // C -> LDS
      }
#pragma unroll
      for (int off = 1; off < 16; off <<= 1) rs += __shfl_xor(rs, off, 32);
      rl[v] = rl[v] * sf[v] + rs;
      rm[v] = newm;
    }
#pragma unroll
    for (int t = 0; t < 4; ++t)
#pragma unroll
      for (int v = 0; v < 8; ++v) o[t][v] *= sf[v];

    // ---- O += P x V  (A re-laid out via per-wave LDS strip) ----
#pragma unroll
    for (int f = 0; f < 2; ++f) {
      Frag16 ap;
      ap.q[0] = *(const uint4*)&P[m16 * LP + 32 * f + 8 * half];
      ap.q[1] = *(const uint4*)&P[m16 * LP + 32 * f + 16 + 8 * half];
#pragma unroll
      for (int t = 0; t < 4; ++t)
        o[t] = __builtin_amdgcn_wmma_f32_16x16x32_bf16(
            false, ap.v, false, vb[f][t].v, (short)0, o[t], false, false);
    }
    __syncthreads();   // all waves done with buffer `cur` before it is refilled
  }

  // ---- normalize and write [B,S,D] fp32 ----
#pragma unroll
  for (int v = 0; v < 8; ++v) {
    const float inv = 1.f / rl[v];
    const int q = q0 + v + 8 * half;
    float* dst = OutAttn + ((size_t)bidx * SEQ + q) * D_MODEL + h * HEAD_DIM;
#pragma unroll
    for (int t = 0; t < 4; ++t) dst[16 * t + m16] = o[t][v] * inv;
  }
}

extern "C" void kernel_launch(void* const* d_in, const int* in_sizes, int n_in,
                              void* d_out, int out_size, void* d_ws, size_t ws_size,
                              hipStream_t stream) {
  const float* x  = (const float*)d_in[0];
  const float* Wq = (const float*)d_in[1];
  const float* bq = (const float*)d_in[2];
  const float* Wk = (const float*)d_in[3];
  const float* bk = (const float*)d_in[4];
  const float* Wv = (const float*)d_in[5];
  const float* bv = (const float*)d_in[6];
  const float* Wo = (const float*)d_in[7];
  const float* bo = (const float*)d_in[8];
  float* out = (float*)d_out;

  char* ws = (char*)d_ws;
  unsigned short* Qb  = (unsigned short*)(ws);                    //  8 MB
  unsigned short* Kbw = (unsigned short*)(ws + (8u  << 20));      //  8 MB
  unsigned short* Vtw = (unsigned short*)(ws + (16u << 20));      //  8 MB
  float*          att = (float*)(ws + (24u << 20));               // 16 MB

  gemm_kernel<0><<<dim3(32, 24), dim3(256), 0, stream>>>(
      x, Wq, Wk, Wv, bq, bk, bv, Qb, Kbw, Vtw, nullptr);
  attn_kernel<<<dim3(SEQ / 128, N_HEADS, 2), dim3(256), 0, stream>>>(
      Qb, Kbw, Vtw, att);
  gemm_kernel<1><<<dim3(32, 8), dim3(256), 0, stream>>>(
      att, Wo, nullptr, nullptr, bo, nullptr, nullptr,
      nullptr, nullptr, nullptr, out);
}
